// BernNet_82231443849681
// MI455X (gfx1250) — compile-verified, matrix operand order checked by reading, not run on hardware
//
#include <hip/hip_runtime.h>
#include <hip/hip_bf16.h>

#define KORD 10

typedef __attribute__((ext_vector_type(16))) __bf16    v16bf;
typedef __attribute__((ext_vector_type(8)))  __bf16    v8bf;
typedef __attribute__((ext_vector_type(8)))  float     v8f;
typedef __attribute__((ext_vector_type(8)))  unsigned  v8u;

__device__ __forceinline__ __bf16 f2bf(float f) {
  unsigned u = __builtin_bit_cast(unsigned, f) + 0x8000u;   // round-to-nearest
  unsigned short h = (unsigned short)(u >> 16);
  return __builtin_bit_cast(__bf16, h);
}

// pack two f32 -> two bf16 in one dword: round (+0x8000) then v_perm_b32
__device__ __forceinline__ unsigned pk2bf(float f0, float f1) {
  unsigned u0 = __builtin_bit_cast(unsigned, f0) + 0x8000u;
  unsigned u1 = __builtin_bit_cast(unsigned, f1) + 0x8000u;
  return __builtin_amdgcn_perm(u1, u0, 0x07060302u);        // {u1.hi16, u0.hi16}
}

// ---------------------------------------------------------------------------
// Repack W [K x Ncols] (row-major f32) into WMMA B-fragment layout (ISA 7.12.2):
//   lane L: n = L%16, half = L/16; element e: k_local = half*8 + (e<8 ? e : e+8)
// Fragment (nt,kt) at ((nt*Ktiles + kt)*32 + lane)*16 elements.
// ---------------------------------------------------------------------------
__global__ void repack_w_kernel(const float* __restrict__ W, __bf16* __restrict__ dst,
                                int Ktiles, int Ntiles, int ldn) {
  int idx = blockIdx.x * blockDim.x + threadIdx.x;
  int total = Ktiles * Ntiles * 512;
  if (idx >= total) return;
  int e    = idx & 15;
  int lane = (idx >> 4) & 31;
  int frag = idx >> 9;
  int kt = frag % Ktiles;
  int nt = frag / Ktiles;
  int n = lane & 15, half = lane >> 4;
  int kl = half * 8 + (e < 8 ? e : e + 8);
  dst[idx] = f2bf(W[(size_t)(kt * 32 + kl) * ldn + nt * 16 + n]);
}

// ---------------------------------------------------------------------------
// GEMM1: h1[N,256] = relu(x[N,512] @ W1 + b1).  Wave owns 16 rows x 256 cols
// (acc[16]); block = 8 waves = 128 rows.  A double-buffered; B loaded in
// chunks of 4 fragments so 4 loads are in flight per WMMA group.
// ---------------------------------------------------------------------------
__global__ __launch_bounds__(256) void gemm1_kernel(
    const float* __restrict__ x, const __bf16* __restrict__ w1f,
    const float* __restrict__ b1, __bf16* __restrict__ h1, int Nn) {
  const int lane = threadIdx.x & 31;
  const int wid  = threadIdx.x >> 5;
  const int rowBase = blockIdx.x * 128 + wid * 16;
  const int m = lane & 15, half = lane >> 4;
  int row = rowBase + m;
  row = row < Nn ? row : Nn - 1;        // clamp: EXEC stays all-ones for WMMA
  const float* xrow = x + (size_t)row * 512;
  const v16bf* btab = (const v16bf*)w1f + lane;

  v8f acc[16] = {};
  // A stage for kt = 0
  const float* ap = xrow + half * 8;
  float4 a0 = *(const float4*)(ap);
  float4 a1 = *(const float4*)(ap + 4);
  float4 a2 = *(const float4*)(ap + 16);
  float4 a3 = *(const float4*)(ap + 20);

  for (int kt = 0; kt < 16; ++kt) {
    v8u ai;
    ai[0] = pk2bf(a0.x, a0.y); ai[1] = pk2bf(a0.z, a0.w);
    ai[2] = pk2bf(a1.x, a1.y); ai[3] = pk2bf(a1.z, a1.w);
    ai[4] = pk2bf(a2.x, a2.y); ai[5] = pk2bf(a2.z, a2.w);
    ai[6] = pk2bf(a3.x, a3.y); ai[7] = pk2bf(a3.z, a3.w);
    v16bf a = __builtin_bit_cast(v16bf, ai);
    if (kt < 15) {                       // issue next A tile early
      const float* np = xrow + (kt + 1) * 32 + half * 8;
      a0 = *(const float4*)(np);
      a1 = *(const float4*)(np + 4);
      a2 = *(const float4*)(np + 16);
      a3 = *(const float4*)(np + 20);
    }
    #pragma unroll
    for (int g = 0; g < 4; ++g) {
      v16bf bb[4];
      #pragma unroll
      for (int t = 0; t < 4; ++t)
        bb[t] = btab[(size_t)((g * 4 + t) * 16 + kt) * 32];
      #pragma unroll
      for (int t = 0; t < 4; ++t)
        acc[g * 4 + t] = __builtin_amdgcn_wmma_f32_16x16x32_bf16(
            false, a, false, bb[t], (short)0, acc[g * 4 + t], false, false);
    }
  }
  const int n = lane & 15, hm = lane >> 4;
  #pragma unroll
  for (int nt = 0; nt < 16; ++nt) {
    const int c = nt * 16 + n;
    const float bias = b1[c];
    #pragma unroll
    for (int v = 0; v < 8; ++v) {
      const int r = rowBase + hm * 8 + v;
      if (r < Nn) h1[(size_t)r * 256 + c] = f2bf(fmaxf(acc[nt][v] + bias, 0.0f));
    }
  }
}

// ---------------------------------------------------------------------------
// GEMM2: h[N,64] = h1[N,256](bf16) @ W2 + b2 (f32 out).  Wave = 16 rows x
// 64 cols (acc[4]); block = 8 waves = 128 rows.  A double-buffered, all 4
// B fragments loaded before the WMMA group.
// ---------------------------------------------------------------------------
__global__ __launch_bounds__(256) void gemm2_kernel(
    const __bf16* __restrict__ h1, const __bf16* __restrict__ w2f,
    const float* __restrict__ b2, float* __restrict__ out, int Nn) {
  const int lane = threadIdx.x & 31;
  const int wid  = threadIdx.x >> 5;
  const int rowBase = blockIdx.x * 128 + wid * 16;
  const int m = lane & 15, half = lane >> 4;
  int row = rowBase + m;
  row = row < Nn ? row : Nn - 1;
  const __bf16* arow = h1 + (size_t)row * 256;
  const v16bf* btab = (const v16bf*)w2f + lane;

  v8f acc[4] = {};
  v8bf lo = *(const v8bf*)(arow + half * 8);
  v8bf hi = *(const v8bf*)(arow + half * 8 + 16);
  for (int kt = 0; kt < 8; ++kt) {
    v16bf a;
    #pragma unroll
    for (int e = 0; e < 8; ++e) { a[e] = lo[e]; a[8 + e] = hi[e]; }
    if (kt < 7) {
      const __bf16* np = arow + (kt + 1) * 32 + half * 8;
      lo = *(const v8bf*)(np);
      hi = *(const v8bf*)(np + 16);
    }
    v16bf bb[4];
    #pragma unroll
    for (int nt = 0; nt < 4; ++nt)
      bb[nt] = btab[(size_t)(nt * 8 + kt) * 32];
    #pragma unroll
    for (int nt = 0; nt < 4; ++nt)
      acc[nt] = __builtin_amdgcn_wmma_f32_16x16x32_bf16(
          false, a, false, bb[nt], (short)0, acc[nt], false, false);
  }
  const int n = lane & 15, hm = lane >> 4;
  #pragma unroll
  for (int nt = 0; nt < 4; ++nt) {
    const int c = nt * 16 + n;
    const float bias = b2[c];
    #pragma unroll
    for (int v = 0; v < 8; ++v) {
      const int r = rowBase + hm * 8 + v;
      if (r < Nn) out[(size_t)r * 64 + c] = acc[nt][v] + bias;
    }
  }
}

// --------------------------- graph propagation -----------------------------
__global__ void fill_kernel(float* __restrict__ p, float v, long n) {
  long i = (long)blockIdx.x * blockDim.x + threadIdx.x;
  if (i < n) p[i] = v;
}

__global__ void copy4_kernel(float* __restrict__ dst, const float* __restrict__ src, long n4) {
  long i = (long)blockIdx.x * blockDim.x + threadIdx.x;
  if (i < n4) ((float4*)dst)[i] = ((const float4*)src)[i];
}

__global__ void deg_kernel(const int* __restrict__ row, float* __restrict__ deg, int Ee) {
  int e = blockIdx.x * blockDim.x + threadIdx.x;
  if (e < Ee) atomicAdd(&deg[row[e]], 1.0f);
}

__global__ void dinv_kernel(const float* __restrict__ deg, float* __restrict__ dinv, int Nn) {
  int i = blockIdx.x * blockDim.x + threadIdx.x;
  if (i < Nn) { float d = deg[i]; dinv[i] = d > 0.0f ? __frsqrt_rn(d) : 0.0f; }
}

__global__ void edgew_kernel(const int* __restrict__ row, const int* __restrict__ col,
                             const float* __restrict__ dinv, float* __restrict__ w, int Ee) {
  int e = blockIdx.x * blockDim.x + threadIdx.x;
  if (e < Ee) w[e] = dinv[row[e]] * dinv[col[e]];
}

// dst[row[e], :] += sign * w[e] * src[col[e], :]
// Wave handles 2 edges (16 lanes x float4 each); edge metadata loaded once by
// lane 0 / lane 16 and broadcast with __shfl (cuts 3 redundant dword loads
// per lane in the atomic-bound stage).
__global__ void scatter_kernel(const int* __restrict__ row, const int* __restrict__ col,
                               const float* __restrict__ w, const float* __restrict__ src,
                               float* __restrict__ dst, float sign, int Ee) {
  const int lane = threadIdx.x & 31;
  long wave = (long)blockIdx.x * 8 + (threadIdx.x >> 5);
  long e = wave * 2 + (lane >> 4);
  if (e >= Ee) return;
  int r = 0, c = 0; float wv = 0.0f;
  if ((lane & 15) == 0) { r = row[e]; c = col[e]; wv = w[e]; }
  const int srcLane = lane & 16;
  r  = __shfl(r,  srcLane);
  c  = __shfl(c,  srcLane);
  wv = __shfl(wv, srcLane);
  wv *= sign;
  const int f = (lane & 15) * 4;
  float4 vv = *(const float4*)(src + (size_t)c * 64 + f);
  float* dp = dst + (size_t)r * 64 + f;
  atomicAdd(dp + 0, wv * vv.x);
  atomicAdd(dp + 1, wv * vv.y);
  atomicAdd(dp + 2, wv * vv.z);
  atomicAdd(dp + 3, wv * vv.w);
}

// dst = coef * src,  coef = binom * relu(temp[k])
__global__ void scale_kernel(float* __restrict__ dst, const float* __restrict__ src,
                             const float* __restrict__ temp, int k, float binom, long n4) {
  long i = (long)blockIdx.x * blockDim.x + threadIdx.x;
  if (i >= n4) return;
  float cc = fmaxf(temp[k], 0.0f) * binom;
  float4 vv = ((const float4*)src)[i];
  vv.x *= cc; vv.y *= cc; vv.z *= cc; vv.w *= cc;
  ((float4*)dst)[i] = vv;
}

// dst = s + coef * v,  coef = binom * relu(temp[k])   (Horner step init)
__global__ void fmacopy_kernel(float* __restrict__ dst, const float* __restrict__ s,
                               const float* __restrict__ v, const float* __restrict__ temp,
                               int k, float binom, long n4) {
  long i = (long)blockIdx.x * blockDim.x + threadIdx.x;
  if (i >= n4) return;
  float cc = fmaxf(temp[k], 0.0f) * binom;
  float4 sv = ((const float4*)s)[i];
  float4 vv = ((const float4*)v)[i];
  sv.x += cc * vv.x; sv.y += cc * vv.y; sv.z += cc * vv.z; sv.w += cc * vv.w;
  ((float4*)dst)[i] = sv;
}

// wave32 per row: log_softmax over 64 classes (2 elements / lane)
__global__ void lsm_kernel(const float* __restrict__ acc, float* __restrict__ out, int Nn) {
  int wid = threadIdx.x >> 5, lane = threadIdx.x & 31;
  int rowi = blockIdx.x * 8 + wid;
  if (rowi >= Nn) return;
  const float* a = acc + (size_t)rowi * 64;
  float x0 = a[lane], x1 = a[lane + 32];
  float mx = fmaxf(x0, x1);
  #pragma unroll
  for (int off = 16; off; off >>= 1) mx = fmaxf(mx, __shfl_xor(mx, off));
  float s = __expf(x0 - mx) + __expf(x1 - mx);
  #pragma unroll
  for (int off = 16; off; off >>= 1) s += __shfl_xor(s, off);
  float lse = mx + __logf(s);
  float* op = out + (size_t)rowi * 64;
  op[lane] = x0 - lse;
  op[lane + 32] = x1 - lse;
}

// ---------------------------------------------------------------------------
extern "C" void kernel_launch(void* const* d_in, const int* in_sizes, int n_in,
                              void* d_out, int out_size, void* d_ws, size_t ws_size,
                              hipStream_t stream) {
  (void)n_in; (void)out_size; (void)ws_size;
  const float* x    = (const float*)d_in[0];
  const int*   ei   = (const int*)d_in[1];
  const float* W1   = (const float*)d_in[2];
  const float* b1   = (const float*)d_in[3];
  const float* W2   = (const float*)d_in[4];
  const float* b2   = (const float*)d_in[5];
  const float* temp = (const float*)d_in[6];

  const int Nn = in_sizes[0] / 512;
  const int Ee = in_sizes[1] / 2;
  const int* row = ei;
  const int* col = ei + Ee;

  char* ws = (char*)d_ws;
  size_t off = 0;
  auto alloc = [&](size_t bytes) { size_t r = off; off += (bytes + 255) & ~(size_t)255; return r; };
  __bf16* w1f  = (__bf16*)(ws + alloc((size_t)512 * 256 * 2));
  __bf16* w2f  = (__bf16*)(ws + alloc((size_t)256 * 64 * 2));
  __bf16* h1   = (__bf16*)(ws + alloc((size_t)Nn * 256 * 2));
  float*  deg  = (float*)(ws + alloc((size_t)Nn * 4));
  float*  dnv  = (float*)(ws + alloc((size_t)Nn * 4));
  float*  wE   = (float*)(ws + alloc((size_t)Ee * 4));
  float*  hbuf = (float*)(ws + alloc((size_t)Nn * 64 * 4));
  float*  vA   = (float*)(ws + alloc((size_t)Nn * 64 * 4));
  float*  vB   = (float*)(ws + alloc((size_t)Nn * 64 * 4));
  float*  SA   = (float*)(ws + alloc((size_t)Nn * 64 * 4));
  float*  SB   = (float*)(ws + alloc((size_t)Nn * 64 * 4));

  // 1) weight repack into WMMA B-fragment layout (bf16)
  repack_w_kernel<<<(512 * 256 + 255) / 256, 256, 0, stream>>>(W1, w1f, 16, 16, 256);
  repack_w_kernel<<<(256 * 64 + 255) / 256, 256, 0, stream>>>(W2, w2f, 8, 4, 64);

  // 2) MLP: x -> relu(xW1+b1) -> (.)W2+b2 = hbuf
  const int gblocks = (Nn + 127) / 128;
  gemm1_kernel<<<gblocks, 256, 0, stream>>>(x, w1f, b1, h1, Nn);
  gemm2_kernel<<<gblocks, 256, 0, stream>>>(h1, w2f, b2, hbuf, Nn);

  // 3) normalized adjacency edge weights
  fill_kernel<<<(Nn + 255) / 256, 256, 0, stream>>>(deg, 0.0f, Nn);
  deg_kernel<<<(Ee + 255) / 256, 256, 0, stream>>>(row, deg, Ee);
  dinv_kernel<<<(Nn + 255) / 256, 256, 0, stream>>>(deg, dnv, Nn);
  edgew_kernel<<<(Ee + 255) / 256, 256, 0, stream>>>(row, col, dnv, wE, Ee);

  const long nfeat = (long)Nn * 64;
  const long n4 = nfeat / 4;
  const int  cb4 = (int)((n4 + 255) / 256);
  const long sct = (long)Ee * 16;
  const int  scb = (int)((sct + 255) / 256);

  // 4) Interleaved Horner:  S = c_K h;  v = h;
  //    for j=1..K:  v <- (I+A)v ;  S <- (I-A)S + c_{K-j} v
  //    (== sum_k c_k (I-A)^k (I+A)^{K-k} h, 20 SpMV passes instead of 65)
  static const float binom[KORD + 1] = {1, 10, 45, 120, 210, 252, 210, 120, 45, 10, 1};
  const float scale = 1.0f / 1024.0f;

  scale_kernel<<<cb4, 256, 0, stream>>>(SA, hbuf, temp, KORD, binom[KORD] * scale, n4);
  float* vCur = hbuf; float* vNxt = vA; float* vOther = vB;
  float* SCur = SA;   float* SNxt = SB;
  for (int j = 1; j <= KORD; ++j) {
    // v <- v + A v
    copy4_kernel<<<cb4, 256, 0, stream>>>(vNxt, vCur, n4);
    scatter_kernel<<<scb, 256, 0, stream>>>(row, col, wE, vCur, vNxt, 1.0f, Ee);
    // S <- S + c_{K-j} * v_new - A S
    fmacopy_kernel<<<cb4, 256, 0, stream>>>(SNxt, SCur, vNxt, temp, KORD - j,
                                            binom[KORD - j] * scale, n4);
    scatter_kernel<<<scb, 256, 0, stream>>>(row, col, wE, SCur, SNxt, -1.0f, Ee);
    // rotate buffers
    float* newCur = vNxt;
    vNxt = (vCur == hbuf) ? vOther : vCur;
    vCur = newCur;
    float* t = SCur; SCur = SNxt; SNxt = t;
  }

  // 5) log_softmax -> d_out
  lsm_kernel<<<(Nn + 7) / 8, 256, 0, stream>>>(SCur, (float*)d_out, Nn);
}
